// RNN_16484084482176
// MI455X (gfx1250) — matrix operand order repaired and checked by reading, not compiled, not run on hardware
//
#include <hip/hip_runtime.h>
#include <stdint.h>

typedef _Float16 half_t;
typedef __attribute__((ext_vector_type(16))) _Float16 v16h;
typedef __attribute__((ext_vector_type(8)))  float    v8f;

#define B_  64
#define S_  256
#define E_  256
#define H_  256
#define L_  1024
#define VL_ 16

// ---------------------------------------------------------------------------
// WMMA helpers (gfx1250 layouts per CDNA5 ISA)
// ---------------------------------------------------------------------------
__device__ __forceinline__ v8f wmma_f16(v16h a, v16h b, v8f c) {
  return __builtin_amdgcn_wmma_f32_16x16x32_f16(false, a, false, b, (short)0, c,
                                                false, false);
}

// A 16x32 f16, row-major source [M,K] (lda = row stride).
__device__ __forceinline__ v16h load_a_frag(const half_t* __restrict__ A, int lda,
                                            int m0, int k0, int lane) {
  const int m  = m0 + (lane & 15);
  const int hi = (lane >> 4) * 16;
  const half_t* p = A + (size_t)m * lda + k0;
  v16h a;
#pragma unroll
  for (int i = 0; i < 8; ++i)  a[i] = p[i + hi];
#pragma unroll
  for (int i = 8; i < 16; ++i) a[i] = p[i + hi];
  return a;
}

// B 32x16 f16, strided gather from row-major [K,N] (GRU only; reloaded from global on every use).
__device__ __forceinline__ v16h load_b_frag(const half_t* __restrict__ Bm, int ldb,
                                            int k0, int n0, int lane) {
  const int n  = n0 + (lane & 15);
  const int hi = (lane >> 4) * 16;
  const half_t* p = Bm + (size_t)(k0 + hi) * ldb + n;
  v16h b;
#pragma unroll
  for (int i = 0; i < 16; ++i) b[i] = p[(size_t)i * ldb];
  return b;
}

// CDNA5 async global->LDS copy (16 bytes/lane) + wait (CDNA5 ISA)
__device__ __forceinline__ void async_copy_b128(uint32_t lds_off, const void* gaddr) {
  asm volatile("global_load_async_to_lds_b128 %0, %1, off"
               :: "v"(lds_off), "v"(gaddr) : "memory");
}
__device__ __forceinline__ void wait_asynccnt_le2() {
  asm volatile("s_wait_asynccnt 0x2" ::: "memory");
}
__device__ __forceinline__ void wait_asynccnt_0() {
  asm volatile("s_wait_asynccnt 0x0" ::: "memory");
}

// ---------------------------------------------------------------------------
// Generic tiled WMMA GEMM: C = act(A[M,K] * Bpacked[K,N] + bias)
// B is pre-packed into native fragment layout: [K/32][N/16][32 lanes][16 halves],
// staged through LDS with double-buffered global_load_async_to_lds_b128.
// block = 256 threads (8 waves), block tile 128x128, wave tile 32x64.
// ACT: 0 = +bias, 1 = relu(+bias), 2 = tanh
// ---------------------------------------------------------------------------
template <int ACT, bool TRANS_STORE, bool OUT_HALF>
__global__ void __launch_bounds__(256)
wmma_gemm_kernel(const half_t* __restrict__ Abase, const half_t* __restrict__ Bbase,
                 const float* __restrict__ bias, void* __restrict__ Cbase,
                 int M, int N, int K, long sA, long sB, long sC) {
  __shared__ half_t bstage[2][8][32][16];  // 16 KB, double buffered

  const int lane = threadIdx.x & 31;
  const int wave = threadIdx.x >> 5;
  const int wm = wave & 3, wn = wave >> 2;
  const int m0 = blockIdx.y * 128 + wm * 32;
  const int n0 = blockIdx.x * 128 + wn * 64;
  const half_t* A  = Abase + (size_t)blockIdx.z * sA;
  const half_t* Bm = Bbase + (size_t)blockIdx.z * sB;
  const int NT = N >> 4;                 // n-tiles in B
  const int NKT = K >> 5;                // k-tiles

  const int cj = threadIdx.x >> 5;       // copy role: tile j = wave id
  const int cl = threadIdx.x & 31;       // copy role: lane slot
  auto stageB = [&](int kt, int buf) {
    const size_t gbyte =
        (((size_t)kt * NT + (size_t)blockIdx.x * 8 + cj) * 32 + cl) * 32;
    const uint32_t lds0 = (uint32_t)(uintptr_t)&bstage[buf][cj][cl][0];
    async_copy_b128(lds0,      (const char*)Bm + gbyte);
    async_copy_b128(lds0 + 16, (const char*)Bm + gbyte + 16);
  };

  v8f acc[2][4] = {};

  stageB(0, 0);
  for (int kt = 0; kt < NKT; ++kt) {
    const int buf = kt & 1;
    const int k0 = kt * 32;
    if (kt + 1 < NKT) {
      stageB(kt + 1, buf ^ 1);   // prefetch next slab into other buffer
      wait_asynccnt_le2();       // current slab (issued first) is complete
    } else {
      wait_asynccnt_0();
    }
    __syncthreads();

    if (k0 + 32 < K)             // global_prefetch_b8 for next A slab
      __builtin_prefetch(A + (size_t)(m0 + (lane & 31)) * K + k0 + 32, 0, 1);

    v16h a0 = load_a_frag(A, K, m0,      k0, lane);
    v16h a1 = load_a_frag(A, K, m0 + 16, k0, lane);
#pragma unroll
    for (int j = 0; j < 4; ++j) {
      v16h b = *(const v16h*)&bstage[buf][wn * 4 + j][lane][0];
      acc[0][j] = wmma_f16(a0, b, acc[0][j]);
      acc[1][j] = wmma_f16(a1, b, acc[1][j]);
    }
    __syncthreads();             // all waves done reading buf before reuse
  }

  const int nn  = lane & 15;
  const int hi8 = (lane >> 4) * 8;
#pragma unroll
  for (int i = 0; i < 2; ++i)
#pragma unroll
    for (int j = 0; j < 4; ++j) {
      const int col = n0 + 16 * j + nn;
      const float bv = (bias != nullptr) ? bias[col] : 0.0f;
#pragma unroll
      for (int r = 0; r < 8; ++r) {
        const int row = m0 + 16 * i + r + hi8;
        float v = acc[i][j][r] + bv;
        if (ACT == 1) v = fmaxf(v, 0.0f);
        if (ACT == 2) v = tanhf(v);
        const size_t idx = TRANS_STORE ? ((size_t)col * M + row)
                                       : ((size_t)row * N + col);
        if (OUT_HALF)
          ((half_t*)Cbase)[(size_t)blockIdx.z * sC + idx] = (half_t)v;
        else
          ((float*)Cbase)[(size_t)blockIdx.z * sC + idx] = v;
      }
    }
}

// ---------------------------------------------------------------------------
// Bidirectional GRU scan. One workgroup per direction, 16 waves.
// h [64,256] double-buffered f16 in LDS; per step gh = h * whhT via WMMA.
// ---------------------------------------------------------------------------
__global__ void __launch_bounds__(512)
gru_scan_kernel(const float* __restrict__ gxF, const float* __restrict__ gxB,
                const half_t* __restrict__ whhTF, const half_t* __restrict__ whhTB,
                const float* __restrict__ bhhF, const float* __restrict__ bhhB,
                half_t* __restrict__ dout) {
  __shared__ half_t hbuf[2][B_][H_];  // 65536 bytes
  const int dir = blockIdx.x;
  const float*  gx   = dir ? gxB   : gxF;
  const half_t* whhT = dir ? whhTB : whhTF;
  const float*  bhh  = dir ? bhhB  : bhhF;

  for (int i = threadIdx.x; i < 2 * B_ * H_; i += blockDim.x)
    (&hbuf[0][0][0])[i] = (half_t)0.0f;
  __syncthreads();

  const int lane = threadIdx.x & 31;
  const int w    = threadIdx.x >> 5;  // 0..15
  const int nn   = lane & 15;
  const int hi8  = (lane >> 4) * 8;
  const int hcol = w * 16 + nn;
  const float bhr = bhh[hcol];
  const float bhz = bhh[H_ + hcol];
  const float bhn = bhh[2 * H_ + hcol];

  for (int step = 0; step < S_; ++step) {
    const int cur = step & 1, nxt = cur ^ 1;
    const int t = dir ? (S_ - 1 - step) : step;
#pragma unroll 1
    for (int m = 0; m < 4; ++m) {  // 4 m-tiles cover B=64
      v8f ar = {}, az = {}, an = {};
#pragma unroll
      for (int k = 0; k < 8; ++k) {  // K = 256
        v16h a   = load_a_frag(&hbuf[cur][0][0], H_, m * 16, k * 32, lane);
        v16h br_ = load_b_frag(whhT, 3 * H_, k * 32, w * 16, lane);
        v16h bz_ = load_b_frag(whhT, 3 * H_, k * 32, H_ + w * 16, lane);
        v16h bn_ = load_b_frag(whhT, 3 * H_, k * 32, 2 * H_ + w * 16, lane);
        ar = wmma_f16(a, br_, ar);
        az = wmma_f16(a, bz_, az);
        an = wmma_f16(a, bn_, an);
      }
#pragma unroll
      for (int r = 0; r < 8; ++r) {
        const int brow = m * 16 + r + hi8;                 // batch index
        const size_t gro = ((size_t)brow * S_ + t) * (3 * H_);
        const float xr = gx[gro + hcol];
        const float xz = gx[gro + H_ + hcol];
        const float xn = gx[gro + 2 * H_ + hcol];
        const float rg = 1.0f / (1.0f + __expf(-(xr + ar[r] + bhr)));
        const float zg = 1.0f / (1.0f + __expf(-(xz + az[r] + bhz)));
        const float ng = tanhf(xn + rg * (an[r] + bhn));
        const float hp = (float)hbuf[cur][brow][hcol];
        const float hv = (1.0f - zg) * ng + zg * hp;
        hbuf[nxt][brow][hcol] = (half_t)hv;
        dout[((size_t)brow * S_ + t) * (2 * H_) + dir * H_ + hcol] = (half_t)hv;
      }
    }
    __syncthreads();
  }
}

// ---------------------------------------------------------------------------
// Packing / utility kernels
// ---------------------------------------------------------------------------
// Pack B = src^T (src f32 [N,K]) into fragment layout [K/32][N/16][32][16] f16.
__global__ void packT_f32_kernel(const float* __restrict__ src,
                                 half_t* __restrict__ dst, int K, int N) {
  const int idx = blockIdx.x * 256 + threadIdx.x;
  if (idx >= K * N) return;
  const int k = idx / N, n = idx % N;
  const int kt = k >> 5, hi = (k >> 4) & 1, i = k & 15;
  const int lane = (n & 15) | (hi << 4), nt = n >> 4;
  dst[(((size_t)kt * (N >> 4) + nt) * 32 + lane) * 16 + i] =
      (half_t)src[(size_t)n * K + k];
}

// Pack B = src (f16 [K,N], batched) into fragment layout.
__global__ void pack_f16_kernel(const half_t* __restrict__ src,
                                half_t* __restrict__ dst, int K, int N, long sElems) {
  const half_t* s = src + (size_t)blockIdx.y * sElems;
  half_t* d = dst + (size_t)blockIdx.y * sElems;
  const int idx = blockIdx.x * 256 + threadIdx.x;
  if (idx >= K * N) return;
  const int k = idx / N, n = idx % N;
  const int kt = k >> 5, hi = (k >> 4) & 1, i = k & 15;
  const int lane = (n & 15) | (hi << 4), nt = n >> 4;
  d[(((size_t)kt * (N >> 4) + nt) * 32 + lane) * 16 + i] = s[idx];
}

__global__ void transpose_convert_kernel(const float* __restrict__ src,
                                         half_t* __restrict__ dst, int R, int C) {
  const int idx = blockIdx.x * blockDim.x + threadIdx.x;
  if (idx >= R * C) return;
  const int r = idx / C, c = idx % C;
  dst[(size_t)c * R + r] = (half_t)src[idx];
}

__global__ void vsum_kernel(const int* __restrict__ Vidx, const float* __restrict__ emb,
                            float* __restrict__ Vsum, float* __restrict__ vnorm) {
  __shared__ float red[256];
  const int l = blockIdx.x, e = threadIdx.x;
  float s = 0.0f;
#pragma unroll
  for (int j = 0; j < VL_; ++j)
    s += emb[(size_t)Vidx[l * VL_ + j] * E_ + e];
  Vsum[(size_t)l * E_ + e] = s;
  red[e] = s * s;
  __syncthreads();
  for (int o = 128; o > 0; o >>= 1) {
    if (e < o) red[e] += red[e + o];
    __syncthreads();
  }
  if (e == 0) vnorm[l] = sqrtf(red[0]);
}

__global__ void embed_kernel(const int* __restrict__ x, const float* __restrict__ emb,
                             half_t* __restrict__ xe) {
  const int row = blockIdx.x, e = threadIdx.x;
  xe[(size_t)row * E_ + e] = (half_t)emb[(size_t)x[row] * E_ + e];
}

__global__ void bn_stats_kernel(const float* __restrict__ src, float* __restrict__ stats,
                                int NB, int NCH) {
  __shared__ float rs[256], rq[256];
  const int ch = blockIdx.x, e = threadIdx.x;
  float s = 0.0f, q = 0.0f;
  for (int b = 0; b < NB; ++b) {
    const float v = src[((size_t)b * NCH + ch) * E_ + e];
    s += v; q += v * v;
  }
  rs[e] = s; rq[e] = q;
  __syncthreads();
  for (int o = 128; o > 0; o >>= 1) {
    if (e < o) { rs[e] += rs[e + o]; rq[e] += rq[e + o]; }
    __syncthreads();
  }
  if (e == 0) {
    const float n = (float)(NB * E_);
    const float mean = rs[0] / n;
    const float var  = rq[0] / n - mean * mean;
    stats[ch * 2]     = mean;
    stats[ch * 2 + 1] = rsqrtf(var + 1e-5f);
  }
}

__global__ void bn_apply_relu_kernel(const float* __restrict__ src,
                                     const float* __restrict__ stats,
                                     const float* __restrict__ g,
                                     const float* __restrict__ be,
                                     half_t* __restrict__ dsth, int chmask) {
  const int row = blockIdx.x, e = threadIdx.x;
  const int ch = row & chmask;
  float v = src[(size_t)row * E_ + e];
  v = g[ch] * (v - stats[ch * 2]) * stats[ch * 2 + 1] + be[ch];
  dsth[(size_t)row * E_ + e] = (half_t)fmaxf(v, 0.0f);
}

__global__ void bn_apply_tanh_kernel(float* __restrict__ buf,
                                     const float* __restrict__ stats,
                                     const float* __restrict__ g,
                                     const float* __restrict__ be, int chmask) {
  const int row = blockIdx.x, e = threadIdx.x;
  const int ch = row & chmask;
  const size_t i = (size_t)row * E_ + e;
  float v = buf[i];
  v = g[ch] * (v - stats[ch * 2]) * stats[ch * 2 + 1] + be[ch];
  buf[i] = tanhf(v);
}

__global__ void cosine_kernel(const float* __restrict__ ebuf,
                              const float* __restrict__ Vsum,
                              const float* __restrict__ vnorm,
                              float* __restrict__ out) {
  __shared__ float rn[256], rq[256];
  const int row = blockIdx.x;            // b*L + l
  const int l = row & (L_ - 1);
  const int e = threadIdx.x;
  const float ev = ebuf[(size_t)row * E_ + e];
  const float vv = Vsum[(size_t)l * E_ + e];
  rn[e] = ev * vv; rq[e] = ev * ev;
  __syncthreads();
  for (int o = 128; o > 0; o >>= 1) {
    if (e < o) { rn[e] += rn[e + o]; rq[e] += rq[e + o]; }
    __syncthreads();
  }
  if (e == 0) {
    const float eps = 1e-8f;
    const float den = fmaxf(sqrtf(rq[0]), eps) * fmaxf(vnorm[l], eps);
    out[row] = rn[0] / den * 10.0f;
  }
}

// ---------------------------------------------------------------------------
// Host launch
// ---------------------------------------------------------------------------
extern "C" void kernel_launch(void* const* d_in, const int* in_sizes, int n_in,
                              void* d_out, int out_size, void* d_ws, size_t ws_size,
                              hipStream_t stream) {
  (void)in_sizes; (void)n_in; (void)out_size; (void)ws_size;
  const int*   x      = (const int*)d_in[0];
  const int*   Vidx   = (const int*)d_in[1];
  const float* emb    = (const float*)d_in[2];
  const float* w_ih_f = (const float*)d_in[3];
  const float* w_hh_f = (const float*)d_in[4];
  const float* b_ih_f = (const float*)d_in[5];
  const float* b_hh_f = (const float*)d_in[6];
  const float* w_ih_b = (const float*)d_in[7];
  const float* w_hh_b = (const float*)d_in[8];
  const float* b_ih_b = (const float*)d_in[9];
  const float* b_hh_b = (const float*)d_in[10];
  const float* w1     = (const float*)d_in[11];
  const float* b1     = (const float*)d_in[12];
  const float* w2     = (const float*)d_in[13];
  const float* b2     = (const float*)d_in[14];
  const float* g1     = (const float*)d_in[15];
  const float* be1    = (const float*)d_in[16];
  const float* g2     = (const float*)d_in[17];
  const float* be2    = (const float*)d_in[18];
  float* out = (float*)d_out;

  char* wsp = (char*)d_ws;
  size_t off = 0;
  auto alloc = [&](size_t bytes) {
    size_t o = off;
    off = (off + bytes + 255) & ~(size_t)255;
    return o;
  };

  half_t* wihPKF = (half_t*)(wsp + alloc((size_t)E_ * 3 * H_ * 2));   // packed [K=E,N=3H]
  half_t* wihPKB = (half_t*)(wsp + alloc((size_t)E_ * 3 * H_ * 2));
  half_t* whhTF  = (half_t*)(wsp + alloc((size_t)H_ * 3 * H_ * 2));   // [H,3H] row-major
  half_t* whhTB  = (half_t*)(wsp + alloc((size_t)H_ * 3 * H_ * 2));
  half_t* w1PK   = (half_t*)(wsp + alloc((size_t)2 * H_ * E_ * 2));   // packed [K=2H,N=E]
  half_t* w2PK   = (half_t*)(wsp + alloc((size_t)2 * H_ * E_ * 2));
  float*  Vsum   = (float*) (wsp + alloc((size_t)L_ * E_ * 4));
  half_t* VsumPK = (half_t*)(wsp + alloc((size_t)E_ * L_ * 2));       // packed [K=E,N=L]
  float*  vnorm  = (float*) (wsp + alloc((size_t)L_ * 4));
  half_t* xe_h   = (half_t*)(wsp + alloc((size_t)B_ * S_ * E_ * 2));
  const size_t gxF_off = alloc((size_t)B_ * S_ * 3 * H_ * 4);
  float*  gxF    = (float*)(wsp + gxF_off);
  float*  gxB    = (float*)(wsp + alloc((size_t)B_ * S_ * 3 * H_ * 4));
  half_t* d_h    = (half_t*)(wsp + alloc((size_t)B_ * S_ * 2 * H_ * 2));
  half_t* d_pk   = (half_t*)(wsp + alloc((size_t)B_ * S_ * 2 * H_ * 2)); // packed per batch
  half_t* d2_h   = (half_t*)(wsp + alloc((size_t)B_ * S_ * E_ * 2));
  half_t* a_h    = (half_t*)(wsp + alloc((size_t)B_ * L_ * S_ * 2));
  half_t* c_h    = (half_t*)(wsp + alloc((size_t)B_ * L_ * 2 * H_ * 2));
  float*  stats1 = (float*)(wsp + alloc((size_t)S_ * 2 * 4));
  float*  stats2 = (float*)(wsp + alloc((size_t)L_ * 2 * 4));
  // Reuse the gx region (dead after the GRU scan) for t1 and t2/e.
  float* t1 = (float*)(wsp + gxF_off);                                // 16.8 MB
  float* t2 = (float*)(wsp + gxF_off + (size_t)B_ * S_ * E_ * 4);     // 67.1 MB

  // 1) pack weights into fragment layout (B = W^T from f32 [N,K] sources)
  auto pkT = [&](const float* src, half_t* dst, int K, int N) {
    const int n = K * N;
    packT_f32_kernel<<<(n + 255) / 256, 256, 0, stream>>>(src, dst, K, N);
  };
  pkT(w_ih_f, wihPKF, E_, 3 * H_);
  pkT(w_ih_b, wihPKB, E_, 3 * H_);
  pkT(w1, w1PK, 2 * H_, E_);
  pkT(w2, w2PK, 2 * H_, E_);
  {  // whh stays row-major [H,3H]: the GRU gathers its fragments from global on every use
    const int n = 3 * H_ * H_;
    transpose_convert_kernel<<<(n + 255) / 256, 256, 0, stream>>>(w_hh_f, whhTF, 3 * H_, H_);
    transpose_convert_kernel<<<(n + 255) / 256, 256, 0, stream>>>(w_hh_b, whhTB, 3 * H_, H_);
  }

  // 2) Vsum + norms, pack Vsum^T, 3) embedding gather
  vsum_kernel<<<L_, 256, 0, stream>>>(Vidx, emb, Vsum, vnorm);
  pkT(Vsum, VsumPK, E_, L_);
  embed_kernel<<<B_ * S_, 256, 0, stream>>>(x, emb, xe_h);

  // 4) gx = xe @ w_ih^T + b_ih (both directions): M=16384, N=768, K=256
  {
    dim3 grid(6, 128, 1);
    wmma_gemm_kernel<0, false, false><<<grid, 256, 0, stream>>>(
        xe_h, wihPKF, b_ih_f, gxF, B_ * S_, 3 * H_, E_, 0, 0, 0);
    wmma_gemm_kernel<0, false, false><<<grid, 256, 0, stream>>>(
        xe_h, wihPKB, b_ih_b, gxB, B_ * S_, 3 * H_, E_, 0, 0, 0);
  }

  // 5) GRU scan, both directions -> d_h [B,S,2H] f16; then pack per batch for B-use
  gru_scan_kernel<<<2, 512, 0, stream>>>(gxF, gxB, whhTF, whhTB, b_hh_f, b_hh_b, d_h);
  {
    dim3 grid((S_ * 2 * H_ + 255) / 256, B_, 1);
    pack_f16_kernel<<<grid, 256, 0, stream>>>(d_h, d_pk, S_, 2 * H_,
                                              (long)S_ * 2 * H_);
  }

  // 6) t1 = d @ w1^T + b1 : M=16384, N=256, K=512 ; BN(channel=s) + relu -> d2_h
  {
    dim3 grid(2, 128, 1);
    wmma_gemm_kernel<0, false, false><<<grid, 256, 0, stream>>>(
        d_h, w1PK, b1, t1, B_ * S_, E_, 2 * H_, 0, 0, 0);
  }
  bn_stats_kernel<<<S_, 256, 0, stream>>>(t1, stats1, B_, S_);
  bn_apply_relu_kernel<<<B_ * S_, 256, 0, stream>>>(t1, stats1, g1, be1, d2_h, S_ - 1);

  // 7) a[b,l,s] = tanh(d2[b] @ Vsum^T) : per-batch M=256, N=1024, K=256, trans store
  {
    dim3 grid(8, 2, B_);
    wmma_gemm_kernel<2, true, true><<<grid, 256, 0, stream>>>(
        d2_h, VsumPK, nullptr, a_h, S_, L_, E_,
        (long)S_ * E_, 0, (long)L_ * S_);
  }

  // 8) c[b] = relu(a[b] @ d[b]) : per-batch M=1024, N=512, K=256
  {
    dim3 grid(4, 8, B_);
    wmma_gemm_kernel<1, false, true><<<grid, 256, 0, stream>>>(
        a_h, d_pk, nullptr, c_h, L_, 2 * H_, S_,
        (long)L_ * S_, (long)S_ * 2 * H_, (long)L_ * 2 * H_);
  }

  // 9) t2 = c @ w2^T + b2 : M=65536, N=256, K=512 ; BN(channel=l) + tanh in place
  {
    dim3 grid(2, 512, 1);
    wmma_gemm_kernel<0, false, false><<<grid, 256, 0, stream>>>(
        c_h, w2PK, b2, t2, B_ * L_, E_, 2 * H_, 0, 0, 0);
  }
  bn_stats_kernel<<<L_, 256, 0, stream>>>(t2, stats2, B_, L_);
  bn_apply_tanh_kernel<<<B_ * L_, 256, 0, stream>>>(t2, stats2, g2, be2, L_ - 1);

  // 10) cosine similarity * 10 -> out [B,L]
  cosine_kernel<<<B_ * L_, 256, 0, stream>>>(t2, Vsum, vnorm, out);
}
